// FlashStarcoderAttention_76562087018612
// MI455X (gfx1250) — matrix operand order repaired and checked by reading, not compiled, not run on hardware
//
#include <hip/hip_runtime.h>

// MI455X (gfx1250) fused StarCoder MQA attention block.
// Stage 1: qkv[f16] = x @ wqkv^T + bqkv   (WMMA f16 GEMM, f32->f16 staging)
// Stage 2: attn[f32] = MQA causal flash attention (WMMA QK^T and PV,
//          K tile staged via global_load_async_to_lds_b128 / ASYNCcnt)
// Stage 3: out[f32] = attn @ wo^T + bo    (WMMA f16 GEMM)

typedef __attribute__((ext_vector_type(16))) _Float16 v16h;
typedef __attribute__((ext_vector_type(8)))  _Float16 v8h;
typedef __attribute__((ext_vector_type(4)))  _Float16 v4h;
typedef __attribute__((ext_vector_type(8)))  float    v8f;

// sched_group_barrier masks
#define SG_WMMA   0x008
#define SG_VMEMRD 0x020
#define SG_DSREAD 0x100

static __device__ __forceinline__ v16h cat8(v8h lo, v8h hi) {
  return __builtin_shufflevector(lo, hi, 0, 1, 2, 3, 4, 5, 6, 7,
                                 8, 9, 10, 11, 12, 13, 14, 15);
}

static __device__ __forceinline__ v4h cvt4(float4 f) {
  v4h h;
  h[0] = (_Float16)f.x; h[1] = (_Float16)f.y;
  h[2] = (_Float16)f.z; h[3] = (_Float16)f.w;
  return h;
}

// ---------------------------------------------------------------------------
// GEMM: C[M,N] = A[M,K](f32) * B[N,K](f32)^T + bias[N], via f16 WMMA.
// Block tile 128x128, 8 waves, each wave 64x32 (4x2 WMMA tiles).
// K-slice of 64 staged through LDS with a register double-buffer pipeline.
// ---------------------------------------------------------------------------
#define GBM 128
#define GBN 128
#define GBK 64
#define GSTR 72  // LDS row stride in halves (64 + 8 pad -> conflict-free b128)

template <typename OutT>
__global__ __launch_bounds__(256) void wmma_gemm_bias(
    const float* __restrict__ A, const float* __restrict__ B,
    const float* __restrict__ bias, OutT* __restrict__ C,
    int M, int N, int K) {
  __shared__ _Float16 lA[GBM * GSTR];
  __shared__ _Float16 lB[GBN * GSTR];

  const int tid   = threadIdx.x;
  const int wave  = tid >> 5;
  const int lane  = tid & 31;
  const int lrow  = lane & 15;
  const int lhalf = lane >> 4;
  const int m0 = blockIdx.y * GBM;
  const int n0 = blockIdx.x * GBN;
  const int wm = (wave >> 2) * 64;  // 2 waves along M
  const int wn = (wave & 3) * 32;   // 4 waves along N

  // Staging decomposition: chunk i covers row (tid>>4)+16*i, cols scol..+3.
  const int srow = tid >> 4;         // 0..15
  const int scol = (tid & 15) << 2;  // 0,4,..,60
  const float* aptr = A + (size_t)(m0 + srow) * K + scol;
  const float* bptr = B + (size_t)(n0 + srow) * K + scol;

  v8f zf = {};
  v8f acc[4][2];
#pragma unroll
  for (int i = 0; i < 4; ++i)
#pragma unroll
    for (int j = 0; j < 2; ++j) acc[i][j] = zf;

  // Prologue: prefetch K-slice 0 into registers.
  float4 ra[8], rb[8];
#pragma unroll
  for (int i = 0; i < 8; ++i) {
    ra[i] = *(const float4*)(aptr + (size_t)(16 * i) * K);
    rb[i] = *(const float4*)(bptr + (size_t)(16 * i) * K);
  }

  for (int k0 = 0; k0 < K; k0 += GBK) {
    // Commit staged registers (f32 -> f16) into LDS as 8B packed stores.
#pragma unroll
    for (int i = 0; i < 8; ++i) {
      *(v4h*)&lA[(srow + 16 * i) * GSTR + scol] = cvt4(ra[i]);
      *(v4h*)&lB[(srow + 16 * i) * GSTR + scol] = cvt4(rb[i]);
    }
    __syncthreads();

    // Issue next slice's global loads now; they overlap the WMMA chain.
    if (k0 + GBK < K) {
#pragma unroll
      for (int i = 0; i < 8; ++i) {
        ra[i] = *(const float4*)(aptr + (size_t)(16 * i) * K + (k0 + GBK));
        rb[i] = *(const float4*)(bptr + (size_t)(16 * i) * K + (k0 + GBK));
      }
    }

#pragma unroll
    for (int ks = 0; ks < 2; ++ks) {
      const int kb = ks * 32;
      // Hoist all fragments so ds_loads issue back-to-back before the WMMAs.
      v16h bf[2], af[4];
#pragma unroll
      for (int j = 0; j < 2; ++j) {
        const _Float16* p = &lB[(wn + j * 16 + lrow) * GSTR + kb + lhalf * 16];
        bf[j] = cat8(*(const v8h*)p, *(const v8h*)(p + 8));
      }
#pragma unroll
      for (int i = 0; i < 4; ++i) {
        const _Float16* p = &lA[(wm + i * 16 + lrow) * GSTR + kb + lhalf * 16];
        af[i] = cat8(*(const v8h*)p, *(const v8h*)(p + 8));
      }
#pragma unroll
      for (int i = 0; i < 4; ++i)
#pragma unroll
        for (int j = 0; j < 2; ++j)
          acc[i][j] = __builtin_amdgcn_wmma_f32_16x16x32_f16(
              false, af[i], false, bf[j], (short)0, acc[i][j], false, false);
      // Pipeline directive: group the 12 ds reads ahead of the 8 WMMAs.
      __builtin_amdgcn_sched_group_barrier(SG_DSREAD, 12, 0);
      __builtin_amdgcn_sched_group_barrier(SG_WMMA, 8, 0);
    }
    __syncthreads();
  }

  // Epilogue: C layout row = r + 8*lhalf, col = lrow.
#pragma unroll
  for (int j = 0; j < 2; ++j) {
    const int col = n0 + wn + j * 16 + lrow;
    const float bv = bias[col];
#pragma unroll
    for (int i = 0; i < 4; ++i) {
      const int row0 = m0 + wm + i * 16 + lhalf * 8;
#pragma unroll
      for (int r = 0; r < 8; ++r)
        C[(size_t)(row0 + r) * N + col] = (OutT)(acc[i][j][r] + bv);
    }
  }
}

// ---------------------------------------------------------------------------
// Multi-query causal flash attention. qkv is [S, 6400] f16:
//   q = cols [head*128, head*128+128), k = cols [6144,6272), v = [6272,6400).
// Block: 4 waves x 32 = 128 threads, 64 query rows of one head.
// K tile staged with CDNA5 async global->LDS DMA (ASYNCcnt-tracked).
// ---------------------------------------------------------------------------
#define AQ 64
#define AK 32
#define KSTR 136  // 128 + 8 pad (halves)
#define VSTR 40   // 32 + 8 pad
#define PSTR 40

__global__ __launch_bounds__(128) void mqa_flash(
    const _Float16* __restrict__ qkv, float* __restrict__ out, int S) {
  constexpr int LD = 6400, HID = 6144, KOFF = 6144, VOFF = 6272;
  __shared__ _Float16 lK[AK * KSTR];     // [key][d]
  __shared__ _Float16 lVT[128 * VSTR];   // [d][key] (transposed)
  __shared__ _Float16 lP[4 * 16 * PSTR]; // per-wave P tile [16][32]

  const int tid   = threadIdx.x;
  const int wave  = tid >> 5;
  const int lane  = tid & 31;
  const int lrow  = lane & 15;
  const int lhalf = lane >> 4;
  const int head  = blockIdx.y;
  const int q0    = blockIdx.x * AQ;
  const int qr    = q0 + wave * 16;  // this wave's first query row
  (void)S;

  // Preload Q fragments for d = 0..127 (4 x 16x32 A-fragments), kept in VGPRs.
  v16h qf[4];
  {
    const _Float16* qb = qkv + (size_t)(qr + lrow) * LD + head * 128;
#pragma unroll
    for (int f = 0; f < 4; ++f) {
      const _Float16* p = qb + f * 32 + lhalf * 16;
      qf[f] = cat8(*(const v8h*)p, *(const v8h*)(p + 8));
    }
  }

  v8f zf = {};
  float rm[8], rl[8];
  v8f acc[8];  // O accumulator: rows r+8*lhalf, d = j*16 + lrow
#pragma unroll
  for (int r = 0; r < 8; ++r) { rm[r] = -3.0e38f; rl[r] = 0.0f; }
#pragma unroll
  for (int j = 0; j < 8; ++j) acc[j] = zf;

  const float sc = 0.08838834764831845f;  // 1/sqrt(128)

  for (int kk = 0; kk < q0 + AQ; kk += AK) {
    // Stage K via async global->LDS DMA (no VGPR round-trip); V via normal
    // loads since the in-register transpose is needed.
    for (int idx = tid; idx < AK * 16; idx += 128) {
      const int r = idx >> 4, c = (idx & 15) << 3;
      const unsigned ldst = (unsigned)(size_t)&lK[r * KSTR + c];
      const unsigned long long gsrc =
          (unsigned long long)(size_t)(qkv + (size_t)(kk + r) * LD + KOFF + c);
      asm volatile("global_load_async_to_lds_b128 %0, %1, off"
                   :: "v"(ldst), "v"(gsrc) : "memory");
      const v8h vv = *(const v8h*)(qkv + (size_t)(kk + r) * LD + VOFF + c);
#pragma unroll
      for (int j = 0; j < 8; ++j) lVT[(c + j) * VSTR + r] = vv[j];
    }
    asm volatile("s_wait_asynccnt 0x0" ::: "memory");
    __syncthreads();

    // S = Q * K^T : two 16x16 tiles; all 8 K-fragments grouped before WMMAs.
    v16h kf[2][4];
#pragma unroll
    for (int t = 0; t < 2; ++t)
#pragma unroll
      for (int f = 0; f < 4; ++f) {
        const _Float16* p = &lK[(t * 16 + lrow) * KSTR + f * 32 + lhalf * 16];
        kf[t][f] = cat8(*(const v8h*)p, *(const v8h*)(p + 8));
      }
    v8f s[2];
    s[0] = zf; s[1] = zf;
#pragma unroll
    for (int t = 0; t < 2; ++t)
#pragma unroll
      for (int f = 0; f < 4; ++f)
        s[t] = __builtin_amdgcn_wmma_f32_16x16x32_f16(
            false, qf[f], false, kf[t][f], (short)0, s[t], false, false);
    __builtin_amdgcn_sched_group_barrier(SG_DSREAD, 16, 0);
    __builtin_amdgcn_sched_group_barrier(SG_WMMA, 8, 0);

    // Online softmax (per-lane rows r+8*lhalf, 16-lane reductions per row).
    float pvv[2][8], ps[8];
#pragma unroll
    for (int r = 0; r < 8; ++r) {
      const int grow = qr + r + lhalf * 8;
#pragma unroll
      for (int t = 0; t < 2; ++t) {
        const int gcol = kk + t * 16 + lrow;
        const float v = s[t][r] * sc;
        pvv[t][r] = (gcol <= grow) ? v : -3.0e38f;
      }
      float mx = fmaxf(pvv[0][r], pvv[1][r]);
#pragma unroll
      for (int d = 1; d < 16; d <<= 1) mx = fmaxf(mx, __shfl_xor(mx, d, 32));
      const float nm = fmaxf(rm[r], mx);
      ps[r] = __expf(rm[r] - nm);
      rm[r] = nm;
      float sum = 0.0f;
#pragma unroll
      for (int t = 0; t < 2; ++t) {
        pvv[t][r] = __expf(pvv[t][r] - nm);
        sum += pvv[t][r];
      }
#pragma unroll
      for (int d = 1; d < 16; d <<= 1) sum += __shfl_xor(sum, d, 32);
      rl[r] = rl[r] * ps[r] + sum;
    }
#pragma unroll
    for (int j = 0; j < 8; ++j)
#pragma unroll
      for (int r = 0; r < 8; ++r) acc[j][r] *= ps[r];

    // Re-layout P (C-layout -> A-fragment) through per-wave LDS.
    _Float16* pw = &lP[wave * 16 * PSTR];
#pragma unroll
    for (int t = 0; t < 2; ++t)
#pragma unroll
      for (int r = 0; r < 8; ++r)
        pw[(r + lhalf * 8) * PSTR + t * 16 + lrow] = (_Float16)pvv[t][r];
    __syncthreads();  // all waves run identical trip counts -> legal

    const _Float16* pp = pw + lrow * PSTR + lhalf * 16;
    const v16h pa = cat8(*(const v8h*)pp, *(const v8h*)(pp + 8));

    // O += P * V : 8 d-tiles; all V fragments grouped ahead of the WMMAs.
    v16h vf[8];
#pragma unroll
    for (int j = 0; j < 8; ++j) {
      const _Float16* vp = &lVT[(j * 16 + lrow) * VSTR + lhalf * 16];
      vf[j] = cat8(*(const v8h*)vp, *(const v8h*)(vp + 8));
    }
#pragma unroll
    for (int j = 0; j < 8; ++j)
      acc[j] = __builtin_amdgcn_wmma_f32_16x16x32_f16(
          false, pa, false, vf[j], (short)0, acc[j], false, false);
    __builtin_amdgcn_sched_group_barrier(SG_DSREAD, 18, 0);
    __builtin_amdgcn_sched_group_barrier(SG_WMMA, 8, 0);
    __syncthreads();
  }

  // Normalize and store f32 [S, HID] at cols head*128 + j*16 + lrow.
#pragma unroll
  for (int r = 0; r < 8; ++r) {
    const float inv = 1.0f / rl[r];
    const size_t base = (size_t)(qr + r + lhalf * 8) * HID + head * 128 + lrow;
#pragma unroll
    for (int j = 0; j < 8; ++j) out[base + j * 16] = acc[j][r] * inv;
  }
}

// ---------------------------------------------------------------------------
extern "C" void kernel_launch(void* const* d_in, const int* in_sizes, int n_in,
                              void* d_out, int out_size, void* d_ws,
                              size_t ws_size, hipStream_t stream) {
  (void)in_sizes; (void)n_in; (void)out_size; (void)ws_size;
  const float* x    = (const float*)d_in[0];
  const float* wqkv = (const float*)d_in[1];
  const float* bqkv = (const float*)d_in[2];
  const float* wo   = (const float*)d_in[3];
  const float* bo   = (const float*)d_in[4];
  float* out = (float*)d_out;

  const int S = 2048, HID = 6144, QKVO = 6400;

  // Workspace: qkv f16 [S,6400] (26.2 MB) + attn f32 [S,6144] (50.3 MB).
  _Float16* qkv = (_Float16*)d_ws;
  float* attn = (float*)((char*)d_ws + (size_t)S * QKVO * sizeof(_Float16));

  dim3 g1(QKVO / GBN, S / GBM);
  wmma_gemm_bias<_Float16><<<g1, 256, 0, stream>>>(x, wqkv, bqkv, qkv,
                                                   S, QKVO, HID);

  dim3 g2(S / AQ, 48);
  mqa_flash<<<g2, 128, 0, stream>>>(qkv, attn, S);

  dim3 g3(HID / GBN, S / GBM);
  wmma_gemm_bias<float><<<g3, 256, 0, stream>>>(attn, wo, bo, out,
                                                S, HID, HID);
}
